// CausalSelfAttention_1056561955425
// MI455X (gfx1250) — matrix operand order repaired and checked by reading, not compiled, not run on hardware
//
#include <hip/hip_runtime.h>
#include <cstddef>
#include <cstdint>

// ---------------- problem constants (match reference) ----------------
constexpr int BATCH = 2;
constexpr int T_SEQ = 2048;
constexpr int C_DIM = 1024;
constexpr int NH    = 16;
constexpr int HD    = 64;          // head dim
constexpr float SOFTMAX_SCALE = 0.125f;   // 1/sqrt(64)
constexpr float NEG_BIG = -3.0e38f;

typedef _Float16 half16 __attribute__((ext_vector_type(16)));
typedef float    floatx8 __attribute__((ext_vector_type(8)));
typedef unsigned int u32x4 __attribute__((ext_vector_type(4)));
typedef int          i32x4 __attribute__((ext_vector_type(4)));
typedef int          i32x8 __attribute__((ext_vector_type(8)));

#if defined(__HIP_DEVICE_COMPILE__) && defined(__has_builtin)
#  if __has_builtin(__builtin_amdgcn_tensor_load_to_lds) && \
      __has_builtin(__builtin_amdgcn_s_wait_tensorcnt)
#    define USE_TDM 1
#  endif
#endif
#ifndef USE_TDM
#  define USE_TDM 0
#endif

// ---------------------------------------------------------------------
// WMMA fragment loaders (CDNA5 wave32 layouts, CDNA5 ISA §7.12.2)
//
// A (16x32 f16): lanes 0-15 -> M=lane, halves[0..7]=K 0..7, halves[8..15]=K 16..23
//                lanes16-31 -> M=lane-16, halves[0..7]=K 8..15, halves[8..15]=K 24..31
// B (32x16 f16): lanes 0-15 -> N=lane,    halves[0..7]=K 0..7, halves[8..15]=K 16..23
//                lanes16-31 -> N=lane-16, halves[0..7]=K 8..15, halves[8..15]=K 24..31
// C (16x16 f32): lanes 0-15 -> N=lane,    float j = row M=j
//                lanes16-31 -> N=lane-16, float j = row M=8+j
// ---------------------------------------------------------------------
__device__ inline half16 load_a16x32(const _Float16* p, int lda, int lane) {
    int row = lane & 15;
    int kb  = (lane >> 4) << 3;            // 0 or 8
    const _Float16* r = p + (size_t)row * lda + kb;
    half16 a;
#pragma unroll
    for (int i = 0; i < 8; ++i) { a[i] = r[i]; a[8 + i] = r[16 + i]; }
    return a;
}

__device__ inline half16 load_b32x16(const _Float16* p, int ldb, int lane) {
    int col = lane & 15;
    int kb  = (lane >> 4) << 3;            // 0 or 8
    const _Float16* r = p + (size_t)col * ldb + kb;
    half16 b;
#pragma unroll
    for (int i = 0; i < 8; ++i) { b[i] = r[i]; b[8 + i] = r[16 + i]; }
    return b;
}

__device__ inline floatx8 wmma_f16(half16 a, half16 b, floatx8 c) {
    return __builtin_amdgcn_wmma_f32_16x16x32_f16(false, a, false, b,
                                                  (short)0, c, false, false);
}

#if USE_TDM
// 1-D f16 DMA copy global -> LDS via the Tensor Data Mover.
// D# bitfields per CDNA5 ISA:
//   group0: count=1 | lds_addr[63:32] | global_addr[120:64] | type=2[127:126]
//   group1: data_size=1(2B)[17:16], tensor_dim0[79:48], tensor_dim1[111:80]=1,
//           tile_dim0[127:112], stride fields don't matter for a single row.
__device__ inline void tdm_copy_f16(uint32_t lds_byte_off, const void* gsrc,
                                    uint32_t nelem /* f16 elements, <= 65535 */) {
    uint64_t ga = (uint64_t)(uintptr_t)gsrc;
    u32x4 g0;
    g0[0] = 1u;                                            // count=1, user desc
    g0[1] = lds_byte_off;                                  // lds_addr
    g0[2] = (uint32_t)(ga & 0xFFFFFFFFu);                  // global_addr[31:0]
    g0[3] = (uint32_t)((ga >> 32) & 0x1FFFFFFu) | (2u << 30); // addr[56:32] | type=2
    i32x8 g1;
#pragma unroll
    for (int i = 0; i < 8; ++i) g1[i] = 0;
    g1[0] = (1 << 16);                                     // data_size = 2 bytes
    g1[1] = (int)((nelem & 0xFFFFu) << 16);                // tensor_dim0[15:0]
    g1[2] = (int)((nelem >> 16) & 0xFFFFu)                 // tensor_dim0[31:16]
          | (1 << 16);                                     // tensor_dim1 = 1
    g1[3] = (int)((nelem & 0xFFFFu) << 16);                // tile_dim0
    g1[5] = (int)nelem;                                    // tensor_dim0_stride lo
    i32x4 z4; z4[0] = z4[1] = z4[2] = z4[3] = 0;
#if __clang_major__ >= 23
    i32x8 z8;
#pragma unroll
    for (int i = 0; i < 8; ++i) z8[i] = 0;
    __builtin_amdgcn_tensor_load_to_lds(g0, g1, z4, z4, z8, 0);
#else
    __builtin_amdgcn_tensor_load_to_lds(g0, g1, z4, z4, 0);
#endif
}
#endif

// ---------------------------------------------------------------------
// Kernel 1: QKV GEMM  [4096,1024] x [1024,3072] + bias -> f16 QKV workspace
// (3 contiguous segments, each [B,H,T,64]). Double-buffered LDS staging:
// fetch tile kt+1 into registers while WMMA consumes LDS buffer kt&1.
// ---------------------------------------------------------------------
__global__ __launch_bounds__(256)
void qkv_gemm_kernel(const float* __restrict__ X, const float* __restrict__ W,
                     const float* __restrict__ bias, _Float16* __restrict__ QKV) {
    constexpr int KD = C_DIM;          // 1024
    constexpr int NC = 3 * C_DIM;      // 3072
    constexpr int NT = KD / 32;        // 32 K-chunks
    __shared__ alignas(32) _Float16 As[2][128][32];   // [buf][m][k]
    __shared__ alignas(32) _Float16 Bs[2][128][32];   // [buf][n][k]

    const int tid  = threadIdx.x;
    const int lane = tid & 31;
    const int wave = tid >> 5;
    const int wm = (wave >> 1) * 32;
    const int wn = (wave & 1) * 64;
    const int m0 = blockIdx.y * 128;
    const int n0 = blockIdx.x * 128;

    // staging thread mapping (coalesced 64B per thread on both tiles)
    const int ar_r  = tid >> 1;
    const int ar_c0 = (tid & 1) * 16;
    const int br_k  = (tid * 16) >> 7;
    const int br_n  = (tid * 16) & 127;
    const float* srcA0 = X + (size_t)(m0 + ar_r) * KD + ar_c0;
    const float* srcB0 = W + (size_t)br_k * NC + n0 + br_n;

    float ar[16], br[16];
    auto fetch = [&](int kt) {
        const float* sa = srcA0 + kt * 32;
        const float* sb = srcB0 + (size_t)kt * 32 * NC;
#pragma unroll
        for (int i = 0; i < 16; ++i) { ar[i] = sa[i]; br[i] = sb[i]; }
    };
    auto commit = [&](int buf) {
#pragma unroll
        for (int i = 0; i < 16; ++i) {
            As[buf][ar_r][ar_c0 + i] = (_Float16)ar[i];
            Bs[buf][br_n + i][br_k]  = (_Float16)br[i];
        }
    };

    floatx8 acc[2][4];
    const floatx8 zero8 = {0.f, 0.f, 0.f, 0.f, 0.f, 0.f, 0.f, 0.f};
#pragma unroll
    for (int i = 0; i < 2; ++i)
#pragma unroll
        for (int j = 0; j < 4; ++j) acc[i][j] = zero8;

    fetch(0);
    commit(0);
    __syncthreads();

    for (int kt = 0; kt < NT; ++kt) {
        const int buf = kt & 1;
        if (kt + 1 < NT) fetch(kt + 1);          // overlap global with WMMA

        half16 a[2], b[4];
#pragma unroll
        for (int mi = 0; mi < 2; ++mi) a[mi] = load_a16x32(&As[buf][wm + mi * 16][0], 32, lane);
#pragma unroll
        for (int ni = 0; ni < 4; ++ni) b[ni] = load_b32x16(&Bs[buf][wn + ni * 16][0], 32, lane);
#pragma unroll
        for (int mi = 0; mi < 2; ++mi)
#pragma unroll
            for (int ni = 0; ni < 4; ++ni)
                acc[mi][ni] = wmma_f16(a[mi], b[ni], acc[mi][ni]);

        if (kt + 1 < NT) commit(buf ^ 1);
        __syncthreads();
    }

    // epilogue: + bias, branch-free scatter into QKV segments
    const size_t seg = (size_t)BATCH * NH * T_SEQ * HD;
    const int rlo = (lane >> 4) << 3;
#pragma unroll
    for (int mi = 0; mi < 2; ++mi) {
#pragma unroll
        for (int j = 0; j < 8; ++j) {
            int row = m0 + wm + mi * 16 + rlo + j;   // 0..4095
            int bb  = row >> 11;
            int t   = row & (T_SEQ - 1);
#pragma unroll
            for (int ni = 0; ni < 4; ++ni) {
                int col = n0 + wn + ni * 16 + (lane & 15);   // 0..3071
                float v = acc[mi][ni][j] + bias[col];
                int which = col >> 10;                        // 0=q 1=k 2=v
                int cc = col & (C_DIM - 1);
                int h  = cc >> 6;
                int hd = cc & (HD - 1);
                size_t idx = (size_t)which * seg +
                             (((size_t)bb * NH + h) * T_SEQ + t) * HD + hd;
                QKV[idx] = (_Float16)v;
            }
        }
    }
}

// ---------------------------------------------------------------------
// Kernel 2: flash attention (causal). block = one (b,h) x 128 q rows.
// 8 waves, wave w owns q rows [q0+16w, q0+16w+16). Key tiles of 64.
// K tile staged by the Tensor Data Mover (contiguous 8KB f16 DMA) when
// available; V tile transposed manually into LDS.
// ---------------------------------------------------------------------
__global__ __launch_bounds__(256)
void attn_kernel(const _Float16* __restrict__ QKV, _Float16* __restrict__ Y) {
    constexpr int QT = 128;
    constexpr int KT = 64;
    const size_t seg = (size_t)BATCH * NH * T_SEQ * HD;

    __shared__ alignas(32) _Float16 Ks[KT][HD];         // [key][hd]   8 KB
    __shared__ alignas(32) _Float16 Vs[HD][KT];         // [hd][key]   8 KB
    __shared__ alignas(32) _Float16 Ps[8][16][KT];      // per-wave P 16 KB

    const int bh   = blockIdx.x;                 // b*NH + h
    const int b    = bh / NH;
    const int h    = bh % NH;
    const int q0   = blockIdx.y * QT;
    const int tid  = threadIdx.x;
    const int lane = tid & 31;
    const int wave = tid >> 5;
    const int qrow0 = q0 + wave * 16;

    const _Float16* Qp = QKV + (size_t)bh * T_SEQ * HD;
    const _Float16* Kp = Qp + seg;
    const _Float16* Vp = Kp + seg;

    // Q A-fragments for this wave's 16 rows (HD=64 -> two K=32 chunks), kept live.
    half16 aq[2];
#pragma unroll
    for (int c = 0; c < 2; ++c)
        aq[c] = load_a16x32(Qp + (size_t)qrow0 * HD + c * 32, HD, lane);

    const floatx8 zero8 = {0.f, 0.f, 0.f, 0.f, 0.f, 0.f, 0.f, 0.f};
    floatx8 o[4];
#pragma unroll
    for (int f = 0; f < 4; ++f) o[f] = zero8;
    float m[8], l[8];
#pragma unroll
    for (int j = 0; j < 8; ++j) { m[j] = NEG_BIG; l[j] = 0.f; }

    const int rlo = (lane >> 4) << 3;
    const int nkt = (q0 + QT) / KT;              // causal key-tile bound (uniform)

#if USE_TDM
    const uint32_t ksOff = (uint32_t)(uintptr_t)(&Ks[0][0]);
#endif

    for (int kt = 0; kt < nkt; ++kt) {
        __syncthreads();                         // prev tile fully consumed
#if USE_TDM
        if (wave == 0)                           // one DMA per block, TENSORcnt
            tdm_copy_f16(ksOff, Kp + (size_t)kt * KT * HD, KT * HD);
#endif
        {
            int r  = tid >> 2;                   // key row 0..63
            int c0 = (tid & 3) * 16;             // hd chunk
            const _Float16* vr = Vp + (size_t)(kt * KT + r) * HD + c0;
#pragma unroll
            for (int i = 0; i < 16; ++i) Vs[c0 + i][r] = vr[i];
#if !USE_TDM
            const _Float16* kr = Kp + (size_t)(kt * KT + r) * HD + c0;
#pragma unroll
            for (int i = 0; i < 16; ++i) Ks[r][c0 + i] = kr[i];
#endif
            if (kt + 1 < nkt) {                  // prefetch next tile
                __builtin_prefetch(vr + (size_t)KT * HD, 0, 1);
                __builtin_prefetch(vr + (size_t)KT * HD - (size_t)seg, 0, 1); // K side
            }
        }
#if USE_TDM
        if (wave == 0) __builtin_amdgcn_s_wait_tensorcnt((short)0);
#endif
        __syncthreads();

        // S = Q * K^T  (scale + causal mask applied per lane)
        floatx8 s[4];
#pragma unroll
        for (int f = 0; f < 4; ++f) {
            floatx8 a = zero8;
#pragma unroll
            for (int c = 0; c < 2; ++c) {
                half16 bk = load_b32x16(&Ks[f * 16][c * 32], HD, lane);
                a = wmma_f16(aq[c], bk, a);
            }
            int colg = kt * KT + f * 16 + (lane & 15);
#pragma unroll
            for (int j = 0; j < 8; ++j) {
                float v = a[j] * SOFTMAX_SCALE;
                if (colg > qrow0 + rlo + j) v = NEG_BIG;
                s[f][j] = v;
            }
        }

        // online softmax (rows 0-7 in lanes 0-15, rows 8-15 in lanes 16-31)
        float newm[8], fac[8], rs[8];
#pragma unroll
        for (int j = 0; j < 8; ++j) {
            float mx = fmaxf(fmaxf(s[0][j], s[1][j]), fmaxf(s[2][j], s[3][j]));
#pragma unroll
            for (int d = 8; d >= 1; d >>= 1) mx = fmaxf(mx, __shfl_xor(mx, d, 32));
            newm[j] = fmaxf(m[j], mx);
            fac[j]  = __expf(m[j] - newm[j]);
            rs[j]   = 0.f;
        }
#pragma unroll
        for (int f = 0; f < 4; ++f) {
#pragma unroll
            for (int j = 0; j < 8; ++j) {
                float p = __expf(s[f][j] - newm[j]);
                rs[j] += p;
                Ps[wave][rlo + j][f * 16 + (lane & 15)] = (_Float16)p;
            }
        }
#pragma unroll
        for (int j = 0; j < 8; ++j) {
            float t = rs[j];
#pragma unroll
            for (int d = 8; d >= 1; d >>= 1) t += __shfl_xor(t, d, 32);
            l[j] = l[j] * fac[j] + t;
            m[j] = newm[j];
        }
#pragma unroll
        for (int f = 0; f < 4; ++f)
#pragma unroll
            for (int j = 0; j < 8; ++j) o[f][j] *= fac[j];

        // same-wave LDS RAW on Ps (DS in-order per wave; fence the compiler too)
        asm volatile("s_wait_dscnt 0" ::: "memory");

        // O += P * V
#pragma unroll
        for (int c = 0; c < 2; ++c) {
            half16 ap = load_a16x32(&Ps[wave][0][c * 32], KT, lane);
#pragma unroll
            for (int f = 0; f < 4; ++f) {
                half16 bv = load_b32x16(&Vs[f * 16][c * 32], KT, lane);
                o[f] = wmma_f16(ap, bv, o[f]);
            }
        }
    }

    // normalize, store y as f16 [B,T,C] (C index = h*64 + hd)
#pragma unroll
    for (int f = 0; f < 4; ++f) {
#pragma unroll
        for (int j = 0; j < 8; ++j) {
            int row = qrow0 + rlo + j;
            int col = h * HD + f * 16 + (lane & 15);
            float v = o[f][j] / l[j];
            Y[((size_t)b * T_SEQ + row) * C_DIM + col] = (_Float16)v;
        }
    }
}

// ---------------------------------------------------------------------
// Kernel 3: output projection  [4096,1024](f16) x [1024,1024] + bias -> f32
// Same double-buffered structure as kernel 1.
// ---------------------------------------------------------------------
__global__ __launch_bounds__(256)
void proj_gemm_kernel(const _Float16* __restrict__ Yh, const float* __restrict__ W,
                      const float* __restrict__ bias, float* __restrict__ Out) {
    constexpr int KD = C_DIM;
    constexpr int NC = C_DIM;
    constexpr int NT = KD / 32;
    __shared__ alignas(32) _Float16 As[2][128][32];
    __shared__ alignas(32) _Float16 Bs[2][128][32];

    const int tid  = threadIdx.x;
    const int lane = tid & 31;
    const int wave = tid >> 5;
    const int wm = (wave >> 1) * 32;
    const int wn = (wave & 1) * 64;
    const int m0 = blockIdx.y * 128;
    const int n0 = blockIdx.x * 128;

    const int ar_r  = tid >> 1;
    const int ar_c0 = (tid & 1) * 16;
    const int br_k  = (tid * 16) >> 7;
    const int br_n  = (tid * 16) & 127;
    const _Float16* srcA0 = Yh + (size_t)(m0 + ar_r) * KD + ar_c0;
    const float*    srcB0 = W + (size_t)br_k * NC + n0 + br_n;

    _Float16 ar[16];
    float    br[16];
    auto fetch = [&](int kt) {
        const _Float16* sa = srcA0 + kt * 32;
        const float*    sb = srcB0 + (size_t)kt * 32 * NC;
#pragma unroll
        for (int i = 0; i < 16; ++i) { ar[i] = sa[i]; br[i] = sb[i]; }
    };
    auto commit = [&](int buf) {
#pragma unroll
        for (int i = 0; i < 16; ++i) {
            As[buf][ar_r][ar_c0 + i] = ar[i];
            Bs[buf][br_n + i][br_k]  = (_Float16)br[i];
        }
    };

    floatx8 acc[2][4];
    const floatx8 zero8 = {0.f, 0.f, 0.f, 0.f, 0.f, 0.f, 0.f, 0.f};
#pragma unroll
    for (int i = 0; i < 2; ++i)
#pragma unroll
        for (int j = 0; j < 4; ++j) acc[i][j] = zero8;

    fetch(0);
    commit(0);
    __syncthreads();

    for (int kt = 0; kt < NT; ++kt) {
        const int buf = kt & 1;
        if (kt + 1 < NT) fetch(kt + 1);

        half16 a[2], b[4];
#pragma unroll
        for (int mi = 0; mi < 2; ++mi) a[mi] = load_a16x32(&As[buf][wm + mi * 16][0], 32, lane);
#pragma unroll
        for (int ni = 0; ni < 4; ++ni) b[ni] = load_b32x16(&Bs[buf][wn + ni * 16][0], 32, lane);
#pragma unroll
        for (int mi = 0; mi < 2; ++mi)
#pragma unroll
            for (int ni = 0; ni < 4; ++ni)
                acc[mi][ni] = wmma_f16(a[mi], b[ni], acc[mi][ni]);

        if (kt + 1 < NT) commit(buf ^ 1);
        __syncthreads();
    }

    const int rlo = (lane >> 4) << 3;
#pragma unroll
    for (int mi = 0; mi < 2; ++mi)
#pragma unroll
        for (int j = 0; j < 8; ++j) {
            int row = m0 + wm + mi * 16 + rlo + j;
#pragma unroll
            for (int ni = 0; ni < 4; ++ni) {
                int col = n0 + wn + ni * 16 + (lane & 15);
                Out[(size_t)row * NC + col] = acc[mi][ni][j] + bias[col];
            }
        }
}

// ---------------------------------------------------------------------
extern "C" void kernel_launch(void* const* d_in, const int* in_sizes, int n_in,
                              void* d_out, int out_size, void* d_ws, size_t ws_size,
                              hipStream_t stream) {
    const float* x      = (const float*)d_in[0];
    const float* w_attn = (const float*)d_in[1];
    const float* b_attn = (const float*)d_in[2];
    const float* w_proj = (const float*)d_in[3];
    const float* b_proj = (const float*)d_in[4];
    float* out = (float*)d_out;

    // workspace: QKV (3 contiguous segments [B,H,T,64] f16) + Y [B,T,C] f16
    const size_t seg = (size_t)BATCH * NH * T_SEQ * HD;   // == B*T*C elements
    _Float16* QKVh = (_Float16*)d_ws;
    _Float16* Yh   = QKVh + 3 * seg;

    dim3 blk(256);
    qkv_gemm_kernel<<<dim3(3 * C_DIM / 128, BATCH * T_SEQ / 128), blk, 0, stream>>>(
        x, w_attn, b_attn, QKVh);
    attn_kernel<<<dim3(BATCH * NH, T_SEQ / 128), blk, 0, stream>>>(QKVh, Yh);
    proj_gemm_kernel<<<dim3(C_DIM / 128, BATCH * T_SEQ / 128), blk, 0, stream>>>(
        Yh, w_proj, b_proj, out);
}
